// Block_9328668967161
// MI455X (gfx1250) — matrix operand order repaired and checked by reading, not compiled, not run on hardware
//
#include <hip/hip_runtime.h>

typedef __attribute__((ext_vector_type(16))) _Float16 v16h;
typedef __attribute__((ext_vector_type(8)))  _Float16 v8h;
typedef __attribute__((ext_vector_type(8)))  float    v8f;
typedef __attribute__((vector_size(16)))     int      v4i;

#define B_   32
#define T_   2048
#define C_   256
#define F_   1024
#define NCH  16
#define ROWS_PER_CH (T_ / NCH)
#define TM   64
#define FC   128            // FF-dimension chunk width
#define EPSF 1e-5f

#if __has_builtin(__builtin_amdgcn_global_load_async_to_lds_b128)
#define HAVE_ASYNC_LDS 1
#else
#define HAVE_ASYNC_LDS 0
#endif

// ---------------- LayerNorm stats over the sequence axis ----------------
__global__ void ln_partial(const float* __restrict__ x,
                           float* __restrict__ psum, float* __restrict__ psq) {
    const int ch = blockIdx.x, b = blockIdx.y, c = threadIdx.x;
    const float* xp = x + ((size_t)b * T_ + (size_t)ch * ROWS_PER_CH) * C_ + c;
    float s = 0.f, q = 0.f;
#pragma unroll 4
    for (int r = 0; r < ROWS_PER_CH; ++r) {
        float v = xp[(size_t)r * C_];
        s += v; q += v * v;
    }
    psum[(b * NCH + ch) * C_ + c] = s;
    psq [(b * NCH + ch) * C_ + c] = q;
}

__global__ void ln_final(const float* __restrict__ psum, const float* __restrict__ psq,
                         float* __restrict__ mu, float* __restrict__ rstd) {
    const int b = blockIdx.x, c = threadIdx.x;
    float s = 0.f, q = 0.f;
#pragma unroll
    for (int ch = 0; ch < NCH; ++ch) {
        s += psum[(b * NCH + ch) * C_ + c];
        q += psq [(b * NCH + ch) * C_ + c];
    }
    const float m   = s / (float)T_;
    const float var = (q - s * m) / (float)(T_ - 1);
    mu  [b * C_ + c] = m;
    rstd[b * C_ + c] = rsqrtf(var + EPSF);
}

// ---------------- Weight downconvert + transpose (one-time, tiny) ----------------
// w1t[f][c] = W1[c][f]  (F x C f16);  w2t[c][f] = W2[f][c]  (C x F f16)
__global__ void prep_weights(const float* __restrict__ W1, const float* __restrict__ W2,
                             _Float16* __restrict__ w1t, _Float16* __restrict__ w2t) {
    const int i = blockIdx.x * blockDim.x + threadIdx.x;   // 0 .. C*F-1
    const int n1 = i / C_, k1 = i % C_;
    w1t[i] = (_Float16)W1[(size_t)k1 * F_ + n1];
    const int n2 = i / F_, k2 = i % F_;
    w2t[i] = (_Float16)W2[(size_t)k2 * C_ + n2];
}

// ---------------- WMMA fragment loaders (contiguous k order, same for A and B) ----------------
// A 16x32 f16: lanes 0-15: row M=lane, K {k0..k0+15};
//              lanes 16-31: row M=lane-16, K {k0+16..k0+31}.
__device__ __forceinline__ v16h load_a16(const _Float16* p, int m0, int k0, int ld, int lane) {
    const _Float16* q = p + (m0 + (lane & 15)) * ld + k0 + ((lane & 16) ? 16 : 0);
    return *(const v16h*)q;
}

// B 32x16 f16 (N-major storage): lanes 0-15: col N=n0+lane, K {k0..k0+15};
// lanes 16-31: col N=n0+lane-16, K {k0+16..k0+31}. One contiguous 32B read/lane.
__device__ __forceinline__ v16h load_b16(const _Float16* w, int n0, int k0, int ldk, int lane) {
    const _Float16* q = w + (n0 + (lane & 15)) * ldk + k0 + ((lane & 16) ? 16 : 0);
    return *(const v16h*)q;
}

// 16-byte global -> LDS copy: async DMA path on CDNA5, direct copy fallback.
__device__ __forceinline__ void cp16(_Float16* ldst, const _Float16* gsrc) {
#if HAVE_ASYNC_LDS
    __builtin_amdgcn_global_load_async_to_lds_b128(
        (__attribute__((address_space(1))) v4i*)(_Float16*)gsrc,
        (__attribute__((address_space(3))) v4i*)ldst, 0, 0);
#else
    *(v8h*)ldst = *(const v8h*)gsrc;
#endif
}

__device__ __forceinline__ void wait_async_copies() {
#if HAVE_ASYNC_LDS
#if __has_builtin(__builtin_amdgcn_s_wait_asynccnt)
    __builtin_amdgcn_s_wait_asynccnt(0);
#else
    asm volatile("s_wait_asynccnt 0x0" ::: "memory");
#endif
#endif
}

// ---------------- Fused LN-apply + FFN + residual ----------------
// One workgroup (8 waves) per 64-row tile of one batch.
__global__ void __launch_bounds__(256) ffn_fused(
    const float* __restrict__ x, const float* __restrict__ gamma, const float* __restrict__ beta,
    const float* __restrict__ b1, const float* __restrict__ b2,
    const _Float16* __restrict__ w1t, const _Float16* __restrict__ w2t,
    const float* __restrict__ mu, const float* __restrict__ rstd,
    float* __restrict__ out)
{
    __shared__ __align__(32) _Float16 hlds [TM * C_];    // 32 KB normalized activations
    __shared__ __align__(32) _Float16 y1lds[TM * FC];    // 16 KB relu(h@W1) chunk
    __shared__ __align__(32) _Float16 w1p  [FC * C_];    // 64 KB W1 chunk panel [FC][C]
    __shared__ __align__(32) _Float16 w2p  [C_ * FC];    // 64 KB W2 chunk panel [C][FC]

    const int tid  = threadIdx.x;
    const int lane = tid & 31;
    const int w    = tid >> 5;            // wave 0..7
    const int b    = blockIdx.y;
    const int t0   = blockIdx.x * TM;

    // Phase A: normalize 64x256 tile into LDS as f16 (lane = channel, coalesced)
    {
        const int   c  = tid;
        const float m  = mu  [b * C_ + c];
        const float rs = rstd[b * C_ + c];
        const float g  = gamma[c];
        const float be = beta [c];
        const float* xp = x + ((size_t)b * T_ + t0) * C_ + c;
#pragma unroll 4
        for (int r = 0; r < TM; ++r)
            hlds[r * C_ + c] = (_Float16)(g * (xp[(size_t)r * C_] - m) * rs + be);
    }
    __syncthreads();

    const int m0  = 16 * (w & 3);         // row block of this wave
    const int grp = w >> 2;               // column group 0/1

    v8f acc2[8];
#pragma unroll
    for (int j = 0; j < 8; ++j) acc2[j] = (v8f){0, 0, 0, 0, 0, 0, 0, 0};

    for (int cc = 0; cc < F_ / FC; ++cc) {
        const int ffb = cc * FC;

        // Stage this chunk's weight panels into LDS (128 KB total, 512 B/thread).
        {
            const _Float16* w1g = w1t + (size_t)ffb * C_;     // contiguous 64 KB slab
#pragma unroll
            for (int q = 0; q < 16; ++q) {
                const int off = (q * 256 + tid) * 8;          // halves
                cp16(&w1p[off], w1g + off);
            }
#pragma unroll
            for (int q = 0; q < 16; ++q) {
                const int idx = q * 256 + tid;                // 4096 x 16B chunks
                const int row = idx >> 4, sub = (idx & 15) * 8;
                cp16(&w2p[row * FC + sub], w2t + (size_t)row * F_ + ffb + sub);
            }
            wait_async_copies();
        }
        __syncthreads();

        // GEMM1 chunk: y1[64 x 128] = relu(h[64 x 256] @ W1panel^T + b1)
        {
            v8f acc1[4];
#pragma unroll
            for (int j = 0; j < 4; ++j) acc1[j] = (v8f){0, 0, 0, 0, 0, 0, 0, 0};
#pragma unroll
            for (int k0 = 0; k0 < C_; k0 += 32) {
                v16h a = load_a16(hlds, m0, k0, C_, lane);    // reused across 4 tiles
#pragma unroll
                for (int j = 0; j < 4; ++j) {
                    v16h bb = load_b16(w1p, grp * 64 + j * 16, k0, C_, lane);
                    acc1[j] = __builtin_amdgcn_wmma_f32_16x16x32_f16(
                        false, a, false, bb, (short)0, acc1[j], false, false);
                }
            }
            const int rsel = (lane & 16) ? 8 : 0;
#pragma unroll
            for (int j = 0; j < 4; ++j) {
                const int   col  = grp * 64 + j * 16 + (lane & 15);
                const float bias = b1[ffb + col];
#pragma unroll
                for (int i = 0; i < 8; ++i) {
                    float v = acc1[j][i] + bias;
                    v = v > 0.f ? v : 0.f;
                    y1lds[(m0 + rsel + i) * FC + col] = (_Float16)v;
                }
            }
        }
        __syncthreads();

        // GEMM2 partial: acc2 += y1[64 x 128] @ W2panel
#pragma unroll
        for (int k0 = 0; k0 < FC; k0 += 32) {
            v16h a = load_a16(y1lds, m0, k0, FC, lane);       // reused across 8 tiles
#pragma unroll
            for (int j = 0; j < 8; ++j) {
                v16h bb = load_b16(w2p, grp * 128 + j * 16, k0, FC, lane);
                acc2[j] = __builtin_amdgcn_wmma_f32_16x16x32_f16(
                    false, a, false, bb, (short)0, acc2[j], false, false);
            }
        }
        __syncthreads();   // protects y1lds + panels before next chunk overwrites
    }

    // Epilogue: + b2 + residual x, write f32 output
    const int rsel = (lane & 16) ? 8 : 0;
#pragma unroll
    for (int j = 0; j < 8; ++j) {
        const int   col  = grp * 128 + j * 16 + (lane & 15);
        const float bias = b2[col];
#pragma unroll
        for (int i = 0; i < 8; ++i) {
            const size_t idx = ((size_t)b * T_ + t0 + m0 + rsel + i) * C_ + col;
            out[idx] = acc2[j][i] + bias + x[idx];
        }
    }
}

extern "C" void kernel_launch(void* const* d_in, const int* in_sizes, int n_in,
                              void* d_out, int out_size, void* d_ws, size_t ws_size,
                              hipStream_t stream) {
    const float* x     = (const float*)d_in[0];
    const float* gamma = (const float*)d_in[1];
    const float* beta  = (const float*)d_in[2];
    const float* W1    = (const float*)d_in[3];
    const float* b1    = (const float*)d_in[4];
    const float* W2    = (const float*)d_in[5];
    const float* b2    = (const float*)d_in[6];
    float* out = (float*)d_out;

    char* ws = (char*)d_ws;
    float* psum = (float*)ws;        ws += (size_t)B_ * NCH * C_ * sizeof(float);
    float* psq  = (float*)ws;        ws += (size_t)B_ * NCH * C_ * sizeof(float);
    float* mu   = (float*)ws;        ws += (size_t)B_ * C_ * sizeof(float);
    float* rstd = (float*)ws;        ws += (size_t)B_ * C_ * sizeof(float);
    _Float16* w1t = (_Float16*)ws;   ws += (size_t)C_ * F_ * sizeof(_Float16);
    _Float16* w2t = (_Float16*)ws;   ws += (size_t)C_ * F_ * sizeof(_Float16);

    ln_partial  <<<dim3(NCH, B_),         256, 0, stream>>>(x, psum, psq);
    prep_weights<<<dim3((C_ * F_) / 256), 256, 0, stream>>>(W1, W2, w1t, w2t);
    ln_final    <<<dim3(B_),              256, 0, stream>>>(psum, psq, mu, rstd);
    ffn_fused   <<<dim3(T_ / TM, B_),     256, 0, stream>>>(x, gamma, beta, b1, b2,
                                                            w1t, w2t, mu, rstd, out);
}
